// Model_16166256902607
// MI455X (gfx1250) — matrix operand order repaired and checked by reading, not compiled, not run on hardware
//
#include <hip/hip_runtime.h>
#include <hip/hip_bf16.h>
#include <stdint.h>

typedef __attribute__((ext_vector_type(16))) _Float16 v16h;
typedef __attribute__((ext_vector_type(8)))  float    v8f;

#define WMMA_F32_F16(a, b, c) \
  __builtin_amdgcn_wmma_f32_16x16x32_f16(false, (a), false, (b), (short)0, (c), false, false)

// ---------------------------------------------------------------------------
// WMMA fragment helpers (wave32, 16x16x32 f16 -> f32)
// A: 16(M) x 32(K) row-major, lane L holds row M=L%16; lanes 0-15 hold K 0-15,
// lanes 16-31 hold K 16-31 (2 halves per VGPR).
// B: 32(K) x 16(N) row-major, lanes 0-15 hold K 0-15, lanes 16-31 hold K 16-31, N=lane%16.
// C/D: v8f, element r -> row M = r + (lane/16)*8, col N = lane%16.
// ---------------------------------------------------------------------------
__device__ __forceinline__ v16h frag_a_f16(const _Float16* A, int lda) {
  const int lane = threadIdx.x & 31;
  const int m = lane & 15;
  const int g = (lane >> 4) & 1;
  v16h a;
#pragma unroll
  for (int v = 0; v < 8; ++v) {
    const int kb = v * 2 + g * 16;
    a[2 * v]     = A[m * lda + kb];
    a[2 * v + 1] = A[m * lda + kb + 1];
  }
  return a;
}

__device__ __forceinline__ v16h frag_b_f16(const _Float16* B, int ldb) {
  const int lane = threadIdx.x & 31;
  const int n = lane & 15;
  const int g = (lane >> 4) & 1;
  v16h b;
#pragma unroll
  for (int i = 0; i < 16; ++i)
    b[i] = B[(g * 16 + i) * ldb + n];
  return b;
}

// B fragment from an N-major tile (rows = N, cols = K, leading dim ldk):
// element i holds B[k = g*16+i][n = lane%16]  ==  tileNK[n][g*16+i]
__device__ __forceinline__ v16h frag_b_from_nk(const _Float16* BNK, int ldk) {
  const int lane = threadIdx.x & 31;
  const int n = lane & 15;
  const int g = (lane >> 4) & 1;
  v16h b;
#pragma unroll
  for (int i = 0; i < 16; ++i)
    b[i] = BNK[n * ldk + g * 16 + i];
  return b;
}

__device__ __forceinline__ float sigm(float x) { return 1.0f / (1.0f + __expf(-x)); }

// Async global -> LDS copy, 16 bytes per lane (GLOBAL_LOAD_ASYNC_TO_LDS_B128,
// tracked by ASYNCcnt). lds_off = low 32 bits of the flat LDS-aperture address.
__device__ __forceinline__ void async_copy_b128(uint32_t lds_off, const void* gaddr) {
  asm volatile("global_load_async_to_lds_b128 %0, %1, off"
               :: "v"(lds_off), "v"(gaddr)
               : "memory");
}
__device__ __forceinline__ void wait_asynccnt0() {
  asm volatile("s_wait_asynccnt 0" ::: "memory");
}

// ===========================================================================
// Kernel 0: f32 -> f16 weight conversion (wih matrices)
// ===========================================================================
__global__ void convert_f16(const float* __restrict__ src, _Float16* __restrict__ dst, int n) {
  const int i = blockIdx.x * blockDim.x + threadIdx.x;
  if (i < n) dst[i] = (_Float16)src[i];
}

// ===========================================================================
// Kernel 1: attention. One block = one (batch, 16-row tile of context).
// Computes s = cdot + qdot + (c*wcq)@q^T + biases (masked), row softmax,
// c2q = a @ q via WMMA. Stores c2q (f32) and per-row max of s (for q2c).
// block = 128 threads (4 waves), grid = (64 tiles, 64 batches)
// ===========================================================================
__global__ void attn_kernel(const float* __restrict__ c, const float* __restrict__ q,
                            const float* __restrict__ c_mask, const float* __restrict__ q_mask,
                            const float* __restrict__ wc, const float* __restrict__ bc,
                            const float* __restrict__ wq, const float* __restrict__ bq,
                            const float* __restrict__ wcq, const float* __restrict__ bcq,
                            float* __restrict__ c2q_out, float* __restrict__ rowmax_out) {
  __shared__ _Float16 qh[128 * 128];   // q tile for this batch, f16  (32 KB)
  __shared__ _Float16 awh[16 * 128];   // (c*wcq) rows, f16           (4 KB)
  __shared__ _Float16 ah[16 * 128];    // softmax(a) rows, f16        (4 KB)
  __shared__ float    stile[16 * 128]; // raw scores                  (8 KB)
  __shared__ float    cdot[16];
  __shared__ float    qdot[128];

  const int b    = blockIdx.y;
  const int row0 = blockIdx.x * 16;
  const int tid  = threadIdx.x;
  const int wave = tid >> 5;
  const int lane = tid & 31;
  const int nl   = lane & 15;
  const int mh   = lane >> 4;

  // stage q (Q_LEN=128, B=64, D=128) -> qh[j][d] f16
  for (int i = tid; i < 128 * 128; i += 128) {
    const int j = i >> 7, d = i & 127;
    qh[i] = (_Float16)q[((size_t)j * 64 + b) * 128 + d];
  }
  { // qdot[j] = q[j] . wq
    const int j = tid;
    float s = 0.f;
    for (int d = 0; d < 128; ++d) s += q[((size_t)j * 64 + b) * 128 + d] * wq[d];
    qdot[j] = s;
  }
  if (tid < 16) { // awh rows + cdot
    const int i = row0 + tid;
    float s = 0.f;
    for (int d = 0; d < 128; ++d) {
      const float cv = c[((size_t)i * 64 + b) * 128 + d];
      awh[tid * 128 + d] = (_Float16)(cv * wcq[d]);
      s += cv * wc[d];
    }
    cdot[tid] = s;
  }
  __syncthreads();

  const float badd = bc[0] + bq[0] + bcq[0];

  // s = awh(16x128) @ qh^T(128x128): each wave does 2 N-tiles of 16
  for (int u = 0; u < 2; ++u) {
    const int nt = wave * 2 + u;
    v8f acc = {};
    for (int kt = 0; kt < 4; ++kt) {
      const v16h a  = frag_a_f16(awh + kt * 32, 128);
      const v16h bb = frag_b_from_nk(qh + (nt * 16) * 128 + kt * 32, 128); // B[k][n]=q[n][k]
      acc = WMMA_F32_F16(a, bb, acc);
    }
#pragma unroll
    for (int r = 0; r < 8; ++r) {
      const int m = r + mh * 8;
      const int j = nt * 16 + nl;
      float v = acc[r] + cdot[m] + qdot[j] + badd;
      if (c_mask[(size_t)b * 1024 + row0 + m] == 0.f) v = -1e30f;
      if (q_mask[(size_t)b * 128 + j] == 0.f) v = -1e30f;
      stile[m * 128 + j] = v;
    }
  }
  __syncthreads();

  // per-row softmax over 128 query positions
  if (tid < 16) {
    float mx = -1e30f;
    for (int j = 0; j < 128; ++j) mx = fmaxf(mx, stile[tid * 128 + j]);
    rowmax_out[(size_t)b * 1024 + row0 + tid] = mx;
    float sum = 0.f;
    for (int j = 0; j < 128; ++j) sum += __expf(stile[tid * 128 + j] - mx);
    const float inv = 1.f / sum;
    for (int j = 0; j < 128; ++j)
      ah[tid * 128 + j] = (_Float16)(__expf(stile[tid * 128 + j] - mx) * inv);
  }
  __syncthreads();

  // c2q = a(16x128) @ q(128x128)
  for (int u = 0; u < 2; ++u) {
    const int nt = wave * 2 + u;
    v8f acc = {};
    for (int kt = 0; kt < 4; ++kt) {
      const v16h a  = frag_a_f16(ah + kt * 32, 128);
      const v16h bb = frag_b_f16(qh + (kt * 32) * 128 + nt * 16, 128);
      acc = WMMA_F32_F16(a, bb, acc);
    }
#pragma unroll
    for (int r = 0; r < 8; ++r) {
      const int m = r + mh * 8;
      c2q_out[((size_t)b * 1024 + row0 + m) * 128 + nt * 16 + nl] = acc[r];
    }
  }
}

// ===========================================================================
// Kernel 2: q2c. One block per batch: softmax over 1024 row-maxes -> bvec,
// q2c[d] = sum_i bvec[i] * c[b][i][d].
// ===========================================================================
__global__ void q2c_kernel(const float* __restrict__ rowmax, const float* __restrict__ c_mask,
                           const float* __restrict__ c, float* __restrict__ q2c_out) {
  const int b = blockIdx.x;
  __shared__ float red[256];
  __shared__ float bv[1024];
  __shared__ float stats[2];
  const int tid = threadIdx.x;

  float mx = -1e30f;
  for (int i = tid; i < 1024; i += 256) {
    const float v = (c_mask[(size_t)b * 1024 + i] == 0.f) ? -1e30f
                                                          : rowmax[(size_t)b * 1024 + i];
    bv[i] = v;
    mx = fmaxf(mx, v);
  }
  red[tid] = mx; __syncthreads();
  for (int s = 128; s > 0; s >>= 1) {
    if (tid < s) red[tid] = fmaxf(red[tid], red[tid + s]);
    __syncthreads();
  }
  if (tid == 0) stats[0] = red[0];
  __syncthreads();
  mx = stats[0];

  float sum = 0.f;
  for (int i = tid; i < 1024; i += 256) {
    const float e = __expf(bv[i] - mx);
    bv[i] = e;
    sum += e;
  }
  __syncthreads();
  red[tid] = sum; __syncthreads();
  for (int s = 128; s > 0; s >>= 1) {
    if (tid < s) red[tid] += red[tid + s];
    __syncthreads();
  }
  if (tid == 0) stats[1] = red[0];
  __syncthreads();
  const float inv = 1.f / stats[1];

  if (tid < 128) {
    const int d = tid;
    float acc = 0.f;
    for (int i = 0; i < 1024; ++i)
      acc += bv[i] * c[((size_t)i * 64 + b) * 128 + d];
    q2c_out[b * 128 + d] = acc * inv;
  }
}

// ===========================================================================
// Kernel 3: assemble g = mask * [cb, c2q, cb*c2q, cb*q2c] -> f16 (T*B, 512)
// ===========================================================================
__global__ void buildg_kernel(const float* __restrict__ c, const float* __restrict__ c2q,
                              const float* __restrict__ q2c, const float* __restrict__ c_mask,
                              _Float16* __restrict__ gh) {
  const int idx = blockIdx.x;          // t*64 + b  (matches c layout (T,B,D))
  const int t = idx >> 6, b = idx & 63;
  const int d = threadIdx.x;           // 0..127
  const float cv = c[(size_t)idx * 128 + d];
  const float c2 = c2q[((size_t)b * 1024 + t) * 128 + d];
  const float qc = q2c[b * 128 + d];
  const float mk = c_mask[(size_t)b * 1024 + t];
  const size_t base = (size_t)idx * 512;
  gh[base + d]       = (_Float16)(mk * cv);
  gh[base + 128 + d] = (_Float16)(mk * c2);
  gh[base + 256 + d] = (_Float16)(mk * cv * c2);
  gh[base + 384 + d] = (_Float16)(mk * cv * qc);
}

// ===========================================================================
// Kernel 4: GEMM  Y(MxN,f32) = X(MxK,f16) @ Wh^T(NxK,f16) + bias
// block 256 thr (8 waves), tile 64x64, K stepped by 32.
// Tiles staged with GLOBAL_LOAD_ASYNC_TO_LDS_B128 (ASYNCcnt), WMMA compute.
// B tile kept N-major in LDS so the async transfers stay contiguous; the
// B fragment is read with a transposed LDS access instead.
// ===========================================================================
__global__ void gemm_xg(const _Float16* __restrict__ X, const _Float16* __restrict__ Wh,
                        const float* __restrict__ bias, float* __restrict__ Y,
                        int M, int N, int K) {
  __shared__ __align__(16) _Float16 Ash[64 * 32];  // [m][k]
  __shared__ __align__(16) _Float16 Bsh[64 * 32];  // [n][k]  (N-major!)
  const int tid  = threadIdx.x;
  const int wave = tid >> 5;
  const int lane = tid & 31;
  const int nl   = lane & 15;
  const int mh   = lane >> 4;
  const int m0 = blockIdx.x * 64;
  const int n0 = blockIdx.y * 64;

  const int t0 = wave * 2, t1 = wave * 2 + 1;        // 16 tiles (4x4) / 8 waves
  const int mt0 = t0 >> 2, nt0 = t0 & 3;
  const int mt1 = t1 >> 2, nt1 = t1 & 3;
  v8f acc0 = {}, acc1 = {};

  const int row = tid >> 2;    // 0..63
  const int seg = tid & 3;     // 4 x 8 halves = 32 halves per row
  const uint32_t ash_off = (uint32_t)(uintptr_t)(Ash + row * 32 + seg * 8);
  const uint32_t bsh_off = (uint32_t)(uintptr_t)(Bsh + row * 32 + seg * 8);

  for (int k0 = 0; k0 < K; k0 += 32) {
    // async stage: 16 contiguous bytes per lane, straight into LDS
    async_copy_b128(ash_off, X  + (size_t)(m0 + row) * K + k0 + seg * 8);
    async_copy_b128(bsh_off, Wh + (size_t)(n0 + row) * K + k0 + seg * 8);
    if (k0 + 32 < K)  // hint next K-slab toward L2 (global_prefetch)
      __builtin_prefetch(X + (size_t)(m0 + row) * K + k0 + 32, 0, 1);
    wait_asynccnt0();
    __syncthreads();
    {
      const v16h a0 = frag_a_f16(Ash + mt0 * 16 * 32, 32);
      const v16h b0 = frag_b_from_nk(Bsh + (nt0 * 16) * 32, 32);
      acc0 = WMMA_F32_F16(a0, b0, acc0);
      const v16h a1 = frag_a_f16(Ash + mt1 * 16 * 32, 32);
      const v16h b1 = frag_b_from_nk(Bsh + (nt1 * 16) * 32, 32);
      acc1 = WMMA_F32_F16(a1, b1, acc1);
    }
    __syncthreads();   // WAR: next slab's async writes must not pass reads
  }
#pragma unroll
  for (int r = 0; r < 8; ++r) {
    const int n = n0 + nt0 * 16 + nl;
    Y[(size_t)(m0 + mt0 * 16 + r + mh * 8) * N + n] = acc0[r] + bias[n];
  }
#pragma unroll
  for (int r = 0; r < 8; ++r) {
    const int n = n0 + nt1 * 16 + nl;
    Y[(size_t)(m0 + mt1 * 16 + r + mh * 8) * N + n] = acc1[r] + bias[n];
  }
}

// ===========================================================================
// Kernel 5: persistent bidirectional LSTM layer. 2 blocks (fwd/bwd), 256 thr.
// Per step: gates = xg[t] + h @ Whh^T (64x256, K=64) via WMMA, activations,
// h/c resident in LDS. Each wave owns two (m-tile, hidden-slice) pairs and their four
// gate tiles i/f/g/o so the nonlinearity folds straight into the accumulators.
// ===========================================================================
__global__ void lstm_layer(const float* __restrict__ xg_f, const float* __restrict__ xg_b,
                           const float* __restrict__ whh_f, const float* __restrict__ whh_b,
                           _Float16* __restrict__ out, int T) {
  const int dir = blockIdx.x;                    // 0 = fwd, 1 = bwd
  const float* __restrict__ xg  = dir ? xg_b  : xg_f;
  const float* __restrict__ whh = dir ? whh_b : whh_f;
  const int col_off = dir * 64;

  __shared__ _Float16 whhT[64 * 256];            // [k][n]   32 KB
  __shared__ _Float16 hh[64 * 64];               // [m][k]    8 KB
  __shared__ float    cc[64 * 64];               //          16 KB
  const int tid  = threadIdx.x;
  const int wave = tid >> 5;
  const int lane = tid & 31;
  const int nl   = lane & 15;
  const int mh   = lane >> 4;

  for (int i = tid; i < 64 * 256; i += 256) {
    const int k = i >> 8, n = i & 255;
    whhT[k * 256 + n] = (_Float16)whh[n * 64 + k];
  }
  for (int i = tid; i < 64 * 64; i += 256) { hh[i] = (_Float16)0.f; cc[i] = 0.f; }
  __syncthreads();

  const int mt = wave >> 1;                      // both tasks of a wave share mt
  for (int s = 0; s < T; ++s) {
    const int t = dir ? (T - 1 - s) : s;
    const size_t xbase = (size_t)t * 64 * 256;

    // read h fragments first, then barrier so no wave overwrites h early
    const v16h a0 = frag_a_f16(hh + mt * 16 * 64, 64);
    const v16h a1 = frag_a_f16(hh + mt * 16 * 64 + 32, 64);
    __syncthreads();

#pragma unroll
    for (int j = 0; j < 2; ++j) {
      const int hb = (wave * 2 + j) & 3;         // hidden slice 0..3
      v8f gi = {}, gf = {}, gg = {}, go = {};
#pragma unroll
      for (int r = 0; r < 8; ++r) {
        const size_t rowx = xbase + (size_t)(mt * 16 + r + mh * 8) * 256;
        const int hcol = hb * 16 + nl;
        gi[r] = xg[rowx + hcol];
        gf[r] = xg[rowx + 64 + hcol];
        gg[r] = xg[rowx + 128 + hcol];
        go[r] = xg[rowx + 192 + hcol];
      }
#pragma unroll
      for (int kt = 0; kt < 2; ++kt) {
        const v16h a = kt ? a1 : a0;
        gi = WMMA_F32_F16(a, frag_b_f16(whhT + kt * 32 * 256 + hb * 16, 256), gi);
        gf = WMMA_F32_F16(a, frag_b_f16(whhT + kt * 32 * 256 + 64 + hb * 16, 256), gf);
        gg = WMMA_F32_F16(a, frag_b_f16(whhT + kt * 32 * 256 + 128 + hb * 16, 256), gg);
        go = WMMA_F32_F16(a, frag_b_f16(whhT + kt * 32 * 256 + 192 + hb * 16, 256), go);
      }
#pragma unroll
      for (int r = 0; r < 8; ++r) {
        const int m   = mt * 16 + r + mh * 8;
        const int hid = hb * 16 + nl;
        const int cell = m * 64 + hid;
        const float cv = sigm(gf[r]) * cc[cell] + sigm(gi[r]) * tanhf(gg[r]);
        cc[cell] = cv;
        const float hv = sigm(go[r]) * tanhf(cv);
        hh[cell] = (_Float16)hv;
        out[((size_t)t * 64 + m) * 128 + col_off + hid] = (_Float16)hv;
      }
    }
    __syncthreads();                             // h writes visible before next step
  }
}

// ===========================================================================
// Kernel 6: logits  p[k][b][t] = [g | m] . pw + pb  (k=0: m2, k=1: r-layer)
// ===========================================================================
__global__ void logits_kernel(const _Float16* __restrict__ gh, const _Float16* __restrict__ m2h,
                              const _Float16* __restrict__ rh,
                              const float* __restrict__ p1w, const float* __restrict__ p1b,
                              const float* __restrict__ p2w, const float* __restrict__ p2b,
                              float* __restrict__ logits) {
  const int idx = blockIdx.x * blockDim.x + threadIdx.x;  // 2 * 65536
  const int k  = idx >> 16;
  const int tb = idx & 65535;                             // t*64 + b
  const _Float16* __restrict__ mrow = (k ? rh : m2h) + (size_t)tb * 128;
  const float* __restrict__ pw = k ? p2w : p1w;
  float acc = k ? p2b[0] : p1b[0];
  const _Float16* __restrict__ grow = gh + (size_t)tb * 512;
  for (int d = 0; d < 512; ++d) acc += (float)grow[d] * pw[d];
  for (int d = 0; d < 128; ++d) acc += (float)mrow[d] * pw[512 + d];
  const int t = tb >> 6, b = tb & 63;
  logits[((size_t)k * 64 + b) * 1024 + t] = acc;
}

// ===========================================================================
// Kernel 7: masked softmax over t, plus the (mask-1)*-1e-30 term. 128 blocks.
// ===========================================================================
__global__ void softmax_out(const float* __restrict__ logits, const float* __restrict__ c_mask,
                            float* __restrict__ out) {
  const int kb = blockIdx.x;       // k*64 + b
  const int b  = kb & 63;
  __shared__ float red[256];
  __shared__ float stats[2];
  const int tid = threadIdx.x;

  float mx = -1e30f;
  for (int t = tid; t < 1024; t += 256) {
    const float v = (c_mask[(size_t)b * 1024 + t] == 0.f) ? -1e30f
                                                          : logits[(size_t)kb * 1024 + t];
    mx = fmaxf(mx, v);
  }
  red[tid] = mx; __syncthreads();
  for (int s = 128; s > 0; s >>= 1) {
    if (tid < s) red[tid] = fmaxf(red[tid], red[tid + s]);
    __syncthreads();
  }
  if (tid == 0) stats[0] = red[0];
  __syncthreads();
  mx = stats[0];

  float sum = 0.f;
  for (int t = tid; t < 1024; t += 256) {
    const float msk = c_mask[(size_t)b * 1024 + t];
    const float v = (msk == 0.f) ? 0.f : __expf(logits[(size_t)kb * 1024 + t] - mx);
    sum += v;
  }
  red[tid] = sum; __syncthreads();
  for (int s = 128; s > 0; s >>= 1) {
    if (tid < s) red[tid] += red[tid + s];
    __syncthreads();
  }
  if (tid == 0) stats[1] = red[0];
  __syncthreads();
  const float inv = 1.f / stats[1];

  for (int t = tid; t < 1024; t += 256) {
    const float msk = c_mask[(size_t)b * 1024 + t];
    const float e = (msk == 0.f) ? 0.f : __expf(logits[(size_t)kb * 1024 + t] - mx);
    out[(size_t)kb * 1024 + t] = e * inv + (msk - 1.f) * -1e-30f;
  }
}

// ===========================================================================
extern "C" void kernel_launch(void* const* d_in, const int* in_sizes, int n_in,
                              void* d_out, int out_size, void* d_ws, size_t ws_size,
                              hipStream_t stream) {
  (void)in_sizes; (void)n_in; (void)out_size; (void)ws_size;
  const float* c      = (const float*)d_in[0];
  const float* q      = (const float*)d_in[1];
  const float* c_mask = (const float*)d_in[2];
  const float* q_mask = (const float*)d_in[3];
  const float* wc     = (const float*)d_in[4];
  const float* bc     = (const float*)d_in[5];
  const float* wq     = (const float*)d_in[6];
  const float* bq     = (const float*)d_in[7];
  const float* wcq    = (const float*)d_in[8];
  const float* bcq    = (const float*)d_in[9];
  const float* m1f_wih = (const float*)d_in[10];
  const float* m1f_whh = (const float*)d_in[11];
  const float* m1f_b   = (const float*)d_in[12];
  const float* m1b_wih = (const float*)d_in[13];
  const float* m1b_whh = (const float*)d_in[14];
  const float* m1b_b   = (const float*)d_in[15];
  const float* m2f_wih = (const float*)d_in[16];
  const float* m2f_whh = (const float*)d_in[17];
  const float* m2f_b   = (const float*)d_in[18];
  const float* m2b_wih = (const float*)d_in[19];
  const float* m2b_whh = (const float*)d_in[20];
  const float* m2b_b   = (const float*)d_in[21];
  const float* rf_wih  = (const float*)d_in[22];
  const float* rf_whh  = (const float*)d_in[23];
  const float* rf_b    = (const float*)d_in[24];
  const float* rb_wih  = (const float*)d_in[25];
  const float* rb_whh  = (const float*)d_in[26];
  const float* rb_b    = (const float*)d_in[27];
  const float* p1_w    = (const float*)d_in[28];
  const float* p1_b    = (const float*)d_in[29];
  const float* p2_w    = (const float*)d_in[30];
  const float* p2_b    = (const float*)d_in[31];

  const int T = 1024, B = 64, M = T * B;   // M = 65536

  // ----- workspace carve-up -----
  char* ws = (char*)d_ws;
  auto carve = [&ws](size_t bytes) {
    char* p = ws;
    ws += (bytes + 255) & ~(size_t)255;
    return p;
  };
  float*    c2q_ws    = (float*)carve((size_t)B * T * 128 * 4);  // 32 MB
  float*    rowmax_ws = (float*)carve((size_t)B * T * 4);        // 256 KB
  float*    q2c_ws    = (float*)carve((size_t)B * 128 * 4);      // 32 KB
  _Float16* gh        = (_Float16*)carve((size_t)M * 512 * 2);   // 64 MB
  float*    xgF       = (float*)carve((size_t)M * 256 * 4);      // 64 MB
  float*    xgB       = (float*)carve((size_t)M * 256 * 4);      // 64 MB
  _Float16* m1h       = (_Float16*)carve((size_t)M * 128 * 2);   // 16 MB
  _Float16* m2h       = (_Float16*)carve((size_t)M * 128 * 2);   // 16 MB
  _Float16* rh        = (_Float16*)carve((size_t)M * 128 * 2);   // 16 MB
  float*    logits_ws = (float*)carve((size_t)2 * B * T * 4);    // 512 KB
  _Float16* m1f_wih_h = (_Float16*)carve((size_t)256 * 512 * 2);
  _Float16* m1b_wih_h = (_Float16*)carve((size_t)256 * 512 * 2);
  _Float16* m2f_wih_h = (_Float16*)carve((size_t)256 * 128 * 2);
  _Float16* m2b_wih_h = (_Float16*)carve((size_t)256 * 128 * 2);
  _Float16* rf_wih_h  = (_Float16*)carve((size_t)256 * 128 * 2);
  _Float16* rb_wih_h  = (_Float16*)carve((size_t)256 * 128 * 2);

  // ----- weight conversion (f32 -> f16) -----
  convert_f16<<<512, 256, 0, stream>>>(m1f_wih, m1f_wih_h, 256 * 512);
  convert_f16<<<512, 256, 0, stream>>>(m1b_wih, m1b_wih_h, 256 * 512);
  convert_f16<<<128, 256, 0, stream>>>(m2f_wih, m2f_wih_h, 256 * 128);
  convert_f16<<<128, 256, 0, stream>>>(m2b_wih, m2b_wih_h, 256 * 128);
  convert_f16<<<128, 256, 0, stream>>>(rf_wih,  rf_wih_h,  256 * 128);
  convert_f16<<<128, 256, 0, stream>>>(rb_wih,  rb_wih_h,  256 * 128);

  // ----- attention flow -----
  attn_kernel<<<dim3(T / 16, B), 128, 0, stream>>>(c, q, c_mask, q_mask, wc, bc, wq, bq,
                                                   wcq, bcq, c2q_ws, rowmax_ws);
  q2c_kernel<<<B, 256, 0, stream>>>(rowmax_ws, c_mask, c, q2c_ws);
  buildg_kernel<<<M, 128, 0, stream>>>(c, c2q_ws, q2c_ws, c_mask, gh);

  // ----- BiLSTM layer 1 (input 512) -----
  gemm_xg<<<dim3(M / 64, 4), 256, 0, stream>>>(gh, m1f_wih_h, m1f_b, xgF, M, 256, 512);
  gemm_xg<<<dim3(M / 64, 4), 256, 0, stream>>>(gh, m1b_wih_h, m1b_b, xgB, M, 256, 512);
  lstm_layer<<<2, 256, 0, stream>>>(xgF, xgB, m1f_whh, m1b_whh, m1h, T);

  // ----- BiLSTM layer 2 (input 128) -----
  gemm_xg<<<dim3(M / 64, 4), 256, 0, stream>>>(m1h, m2f_wih_h, m2f_b, xgF, M, 256, 128);
  gemm_xg<<<dim3(M / 64, 4), 256, 0, stream>>>(m1h, m2b_wih_h, m2b_b, xgB, M, 256, 128);
  lstm_layer<<<2, 256, 0, stream>>>(xgF, xgB, m2f_whh, m2b_whh, m2h, T);

  // ----- BiLSTM layer r (input 128) -----
  gemm_xg<<<dim3(M / 64, 4), 256, 0, stream>>>(m2h, rf_wih_h, rf_b, xgF, M, 256, 128);
  gemm_xg<<<dim3(M / 64, 4), 256, 0, stream>>>(m2h, rb_wih_h, rb_b, xgB, M, 256, 128);
  lstm_layer<<<2, 256, 0, stream>>>(xgF, xgB, rf_whh, rb_whh, rh, T);

  // ----- pointers + softmax -----
  logits_kernel<<<2 * M / 256, 256, 0, stream>>>(gh, m2h, rh, p1_w, p1_b, p2_w, p2_b,
                                                 logits_ws);
  softmax_out<<<2 * B, 256, 0, stream>>>(logits_ws, c_mask, (float*)d_out);
}
